// GraphTransformer_7206955123257
// MI455X (gfx1250) — matrix operand order repaired and checked by reading, not compiled, not run on hardware
//
#include <hip/hip_runtime.h>
#include <hip/hip_bf16.h>
#include <math.h>

// ---------------------------------------------------------------------------
// Graph transformer for MI455X (gfx1250, wave32).
//  - Dense node GEMMs on the WMMA pipe: v_wmma_f32_16x16x32_f16, fp32
//    accumulate, f16 operands converted in-register (data stays fp32 in HBM).
//    Each wave computes a 16x64 strip: A tile converted once, 4 WMMAs reuse it;
//    B read from LDS (transposed layout) as 32B vectors; K loop fully unrolled.
//  - Edge attention phase is bandwidth/atomic bound -> scalar VALU kernels,
//    rank-1 edge embedding folded in (never materialized).
//  - LayerNorm/head-mean: one wave32 per node, shfl_xor reductions.
// ---------------------------------------------------------------------------

typedef __attribute__((ext_vector_type(16))) _Float16 v16h;
typedef __attribute__((ext_vector_type(8)))  float    v8f;
typedef __attribute__((ext_vector_type(4)))  float    f32x4;

#define NODES 50000
#define EDGES 160000
#define HD    256      // H*D
#define DD    64

// ---------------- WMMA GEMM: Y[M,N] = act(A[M,K] @ W[K,N] + b) --------------
// One wave computes a 16x(16*NT) strip. W staged transposed in LDS as f16.
// A layout (ISA 7.12.2, 16-bit A 16x32): lane l -> row m=l&15;
//   hs = l>>4: elems 0..7 -> K = 8*hs + j ; elems 8..15 -> K = 16 + 8*hs + j
// B layout (32x16): lane l -> col n=l&15; same K order as A
// C/D layout: lane l -> col n=l&15; vgpr r -> row m = 8*(l>>4) + r
template <int K, int NT>
__global__ __launch_bounds__(256) void wmma_gemm_kernel(
    const float* __restrict__ A, const float* __restrict__ W,
    const float* __restrict__ bias, float* __restrict__ Y,
    int M, int N, int act)
{
    __shared__ _Float16 sWt[256 * 64];          // [n][k] transposed, 32 KB of 320 KB LDS
    int tid = threadIdx.x;
    int total = K * N;
    for (int i = tid; i < total; i += 256) {
        int kk = i / N, nn = i - kk * N;
        sWt[nn * K + kk] = (_Float16)W[i];
    }
    __syncthreads();

    int wave = tid >> 5;
    int lane = tid & 31;
    int nGrpN  = N / (16 * NT);
    int nTiles = (M >> 4) * nGrpN;
    int tileIdx = blockIdx.x * 8 + wave;        // wave-uniform -> EXEC stays all-1s
    if (tileIdx >= nTiles) return;

    int tileM = tileIdx / nGrpN;
    int grpN  = tileIdx - tileM * nGrpN;
    int row0 = tileM << 4;
    int n0   = grpN * (16 * NT);

    int ml = lane & 15;
    int hs = lane >> 4;
    const float* arow = A + (size_t)(row0 + ml) * K;
    int kbA = hs * 8;
    int khB = hs * 8;

    v8f c[NT];
#pragma unroll
    for (int t = 0; t < NT; ++t) c[t] = (v8f){};

#pragma unroll
    for (int k0 = 0; k0 < K; k0 += 32) {
        // A tile: 16 halfs/lane, fetched as 4x b128 (all offsets 16B aligned)
        f32x4 a0 = *(const f32x4*)(arow + k0 + kbA);
        f32x4 a1 = *(const f32x4*)(arow + k0 + kbA + 4);
        f32x4 a2 = *(const f32x4*)(arow + k0 + 16 + kbA);
        f32x4 a3 = *(const f32x4*)(arow + k0 + 16 + kbA + 4);
        v16h a;
#pragma unroll
        for (int j = 0; j < 4; ++j) {
            a[j]      = (_Float16)a0[j];
            a[j + 4]  = (_Float16)a1[j];
            a[j + 8]  = (_Float16)a2[j];
            a[j + 12] = (_Float16)a3[j];
        }
#pragma unroll
        for (int t = 0; t < NT; ++t) {
            int n = n0 + t * 16 + ml;
            typedef __attribute__((ext_vector_type(8))) _Float16 v8h_t;
            v8h_t b0 = *(const v8h_t*)(sWt + n * K + k0 + khB);
            v8h_t b1 = *(const v8h_t*)(sWt + n * K + k0 + 16 + khB);
            v16h b = __builtin_shufflevector(b0, b1, 0,1,2,3,4,5,6,7,8,9,10,11,12,13,14,15);  // 32B, 2x ds_load_b128
            c[t] = __builtin_amdgcn_wmma_f32_16x16x32_f16(
                       false, a, false, b, (short)0, c[t], false, false);
        }
    }

    int mb = hs * 8;
#pragma unroll
    for (int t = 0; t < NT; ++t) {
        int n = n0 + t * 16 + ml;
        float bv = bias ? bias[n] : 0.0f;
#pragma unroll
        for (int r = 0; r < 8; ++r) {
            float v = c[t][r] + bv;
            if (act) v = (v >= 0.0f) ? v : 0.01f * v;
            Y[(size_t)(row0 + mb + r) * N + n] = v;
        }
    }
}

// ---------------- GEMV: y[M] = A[M,64] @ w[64] + b (wave per row) ----------
__global__ void gemv64_kernel(const float* __restrict__ A, const float* __restrict__ w,
                              const float* __restrict__ bias, float* __restrict__ y, int M)
{
    int gid  = blockIdx.x * blockDim.x + threadIdx.x;
    int wid  = gid >> 5;
    int lane = threadIdx.x & 31;
    if (wid >= M) return;
    const float* a = A + (size_t)wid * 64;
    float s = a[lane] * w[lane] + a[lane + 32] * w[lane + 32];
    for (int off = 16; off; off >>= 1) s += __shfl_xor(s, off, 32);
    if (lane == 0) y[wid] = s + bias[0];
}

// ---------------- edge attention ------------------------------------------
__device__ inline unsigned f2ord(float f) {
    unsigned u = __float_as_uint(f);
    return (u & 0x80000000u) ? ~u : (u | 0x80000000u);
}
__device__ inline float ord2f(unsigned u) {
    unsigned v = (u & 0x80000000u) ? (u & 0x7FFFFFFFu) : ~u;
    return __uint_as_float(v);
}

// alpha[e,h] = (q[dst,h,:] . (k[src,h,:] + ea*We[h,:])) / 8 ; segment max via atomics
__global__ void edge_alpha_kernel(const float* __restrict__ q, const float* __restrict__ k,
    const float* __restrict__ ea_arr, const float* __restrict__ We,
    const int* __restrict__ src, const int* __restrict__ dst,
    float* __restrict__ alphabuf, unsigned* __restrict__ amax, int E)
{
    int t = blockIdx.x * blockDim.x + threadIdx.x;
    if (t >= E * 4) return;
    int eid = t >> 2, h = t & 3;
    int s = src[eid], d = dst[eid];
    float ea = ea_arr[eid];
    const float* qp = q + (size_t)d * HD + h * DD;
    const float* kp = k + (size_t)s * HD + h * DD;
    const float* wp = We + h * DD;
    float dot = 0.0f;
#pragma unroll 8
    for (int i = 0; i < DD; ++i) dot += qp[i] * (kp[i] + ea * wp[i]);
    float alpha = dot * 0.125f;   // 1/sqrt(64)
    alphabuf[t] = alpha;
    atomicMax(&amax[d * 4 + h], f2ord(alpha));
}

__global__ void edge_exp_kernel(float* __restrict__ alphabuf, const unsigned* __restrict__ amax,
                                const int* __restrict__ dst, float* __restrict__ denom, int E)
{
    int t = blockIdx.x * blockDim.x + threadIdx.x;
    if (t >= E * 4) return;
    int eid = t >> 2, h = t & 3;
    int d = dst[eid];
    float a = __expf(alphabuf[t] - ord2f(amax[d * 4 + h]));
    alphabuf[t] = a;
    atomicAdd(&denom[d * 4 + h], a);
}

__global__ void edge_msg_kernel(const float* __restrict__ v, const float* __restrict__ ea_arr,
    const float* __restrict__ We, const float* __restrict__ alphabuf,
    const float* __restrict__ denom, const int* __restrict__ src, const int* __restrict__ dst,
    float* __restrict__ agg, int E)
{
    int t = blockIdx.x * blockDim.x + threadIdx.x;
    if (t >= E * 4) return;
    int eid = t >> 2, h = t & 3;
    int s = src[eid], d = dst[eid];
    float ea = ea_arr[eid];
    float a = alphabuf[t] / (denom[d * 4 + h] + 1e-16f);
    const float* vp = v + (size_t)s * HD + h * DD;
    const float* wp = We + h * DD;
    float* ap = agg + (size_t)d * HD + h * DD;
#pragma unroll 8
    for (int i = 0; i < DD; ++i) atomicAdd(&ap[i], (vp[i] + ea * wp[i]) * a);
}

// ---------------- node LN kernels (wave per node, 2 channels/lane) ----------
// x += LN( mean_h(agg) + skip ; g,c )
__global__ void node_merge_ln_kernel(float* __restrict__ x, const float* __restrict__ agg,
    const float* __restrict__ skip, const float* __restrict__ g, const float* __restrict__ c,
    int Nn)
{
    int gid  = blockIdx.x * blockDim.x + threadIdx.x;
    int wid  = gid >> 5;
    int lane = threadIdx.x & 31;
    if (wid >= Nn) return;
    const float* ag = agg + (size_t)wid * HD;
    float t0 = 0.0f, t1 = 0.0f;
#pragma unroll
    for (int h = 0; h < 4; ++h) { t0 += ag[h * DD + lane]; t1 += ag[h * DD + 32 + lane]; }
    t0 = t0 * 0.25f + skip[(size_t)wid * DD + lane];
    t1 = t1 * 0.25f + skip[(size_t)wid * DD + 32 + lane];
    float s = t0 + t1;
    for (int off = 16; off; off >>= 1) s += __shfl_xor(s, off, 32);
    float mean = s * (1.0f / 64.0f);
    float d0 = t0 - mean, d1 = t1 - mean;
    float vs = d0 * d0 + d1 * d1;
    for (int off = 16; off; off >>= 1) vs += __shfl_xor(vs, off, 32);
    float rstd = rsqrtf(vs * (1.0f / 64.0f) + 1e-5f);
    x[(size_t)wid * DD + lane]      += d0 * rstd * g[lane]      + c[lane];
    x[(size_t)wid * DD + 32 + lane] += d1 * rstd * g[32 + lane] + c[32 + lane];
}

// x += LN( y ; g,c )
__global__ void node_ln_add_kernel(float* __restrict__ x, const float* __restrict__ y,
    const float* __restrict__ g, const float* __restrict__ c, int Nn)
{
    int gid  = blockIdx.x * blockDim.x + threadIdx.x;
    int wid  = gid >> 5;
    int lane = threadIdx.x & 31;
    if (wid >= Nn) return;
    float t0 = y[(size_t)wid * DD + lane];
    float t1 = y[(size_t)wid * DD + 32 + lane];
    float s = t0 + t1;
    for (int off = 16; off; off >>= 1) s += __shfl_xor(s, off, 32);
    float mean = s * (1.0f / 64.0f);
    float d0 = t0 - mean, d1 = t1 - mean;
    float vs = d0 * d0 + d1 * d1;
    for (int off = 16; off; off >>= 1) vs += __shfl_xor(vs, off, 32);
    float rstd = rsqrtf(vs * (1.0f / 64.0f) + 1e-5f);
    x[(size_t)wid * DD + lane]      += d0 * rstd * g[lane]      + c[lane];
    x[(size_t)wid * DD + 32 + lane] += d1 * rstd * g[32 + lane] + c[32 + lane];
}

// ---------------- utility ---------------------------------------------------
__global__ void fill_f32_kernel(float* p, float v, int n) {
    int i = blockIdx.x * blockDim.x + threadIdx.x;
    if (i < n) p[i] = v;
}
__global__ void fill_u32_kernel(unsigned* p, unsigned v, int n) {
    int i = blockIdx.x * blockDim.x + threadIdx.x;
    if (i < n) p[i] = v;
}
__global__ void reduce_sum_kernel(const float* __restrict__ in, float* __restrict__ out, int n) {
    __shared__ float sh[256];
    float s = 0.0f;
    for (int i = blockIdx.x * blockDim.x + threadIdx.x; i < n; i += gridDim.x * blockDim.x)
        s += in[i];
    sh[threadIdx.x] = s;
    __syncthreads();
    for (int off = 128; off; off >>= 1) {
        if (threadIdx.x < (unsigned)off) sh[threadIdx.x] += sh[threadIdx.x + off];
        __syncthreads();
    }
    if (threadIdx.x == 0) atomicAdd(out, sh[0]);
}

// ---------------- host side -------------------------------------------------
static inline void launch_gemm(const float* A, const float* W, const float* b, float* Y,
                               int M, int K, int N, int act, hipStream_t s)
{
    int nGrp   = N / 64;                    // NT=4 -> 64 cols per wave
    int tiles  = (M >> 4) * nGrp;
    int blocks = (tiles + 7) / 8;
    if (K == 32)
        wmma_gemm_kernel<32, 4><<<blocks, 256, 0, s>>>(A, W, b, Y, M, N, act);
    else
        wmma_gemm_kernel<64, 4><<<blocks, 256, 0, s>>>(A, W, b, Y, M, N, act);
}

extern "C" void kernel_launch(void* const* d_in, const int* in_sizes, int n_in,
                              void* d_out, int out_size, void* d_ws, size_t ws_size,
                              hipStream_t stream)
{
    // Input flattening (insertion order of setup_inputs pytree):
    // 0:x 1:edge_attr
    // 2..7:   pre  W0,b0,W1,b1,W2,b2
    // 8..22:  layer0 Wq,bq,Wk,bk,Wv,bv,We,Wskip,bskip,Wlin,blin,g0,c0,g1,c1
    // 23..37: layer1 (same)
    // 38..43: charge  44..49: G  50..55: gap
    // 56: edge_index  57: batch
    auto F = [&](int i) { return (const float*)d_in[i]; };
    const float* Xin = F(0);
    const float* EA  = F(1);
    const int*   EI  = (const int*)d_in[56];
    const int* src = EI;
    const int* dst = EI + EDGES;

    // workspace layout (floats)
    float* ws = (float*)d_ws;
    size_t o = 0;
    float* X    = ws + o; o += (size_t)NODES * DD;
    float* T1   = ws + o; o += (size_t)NODES * DD;
    float* T2   = ws + o; o += (size_t)NODES * DD;
    float* SKIP = ws + o; o += (size_t)NODES * DD;
    float* Q    = ws + o; o += (size_t)NODES * HD;
    float* Kb   = ws + o; o += (size_t)NODES * HD;
    float* Vb   = ws + o; o += (size_t)NODES * HD;
    float* AGG  = ws + o; o += (size_t)NODES * HD;
    float* ALPH = ws + o; o += (size_t)EDGES * 4;
    float* DEN  = ws + o; o += (size_t)NODES * 4;
    float* GB   = ws + o; o += (size_t)NODES;
    unsigned* AMAX = (unsigned*)(ws + o); o += (size_t)NODES * 4;

    float* out = (float*)d_out;   // [0]=G, [1]=gap, [2..]=charge

    const int EH   = EDGES * 4;
    const int ebk  = (EH + 255) / 256;
    const int nwbk = (NODES + 7) / 8;            // wave-per-node, 8 waves/block
    const int aggN = NODES * HD;

    // ---- pre MLP: 32->64->64->64
    launch_gemm(Xin, F(2), F(3), T1, NODES, 32, 64, 1, stream);
    launch_gemm(T1,  F(4), F(5), T2, NODES, 64, 64, 1, stream);
    launch_gemm(T2,  F(6), F(7), X,  NODES, 64, 64, 0, stream);

    // ---- transformer layers
    for (int l = 0; l < 2; ++l) {
        int p = 8 + l * 15;
        launch_gemm(X, F(p + 0), F(p + 1), Q,    NODES, 64, 256, 0, stream);   // Wq,bq
        launch_gemm(X, F(p + 2), F(p + 3), Kb,   NODES, 64, 256, 0, stream);   // Wk,bk
        launch_gemm(X, F(p + 4), F(p + 5), Vb,   NODES, 64, 256, 0, stream);   // Wv,bv
        launch_gemm(X, F(p + 7), F(p + 8), SKIP, NODES, 64, 64,  0, stream);   // Wskip,bskip
        const float* We = F(p + 6);

        fill_u32_kernel<<<(NODES * 4 + 255) / 256, 256, 0, stream>>>(AMAX, 0u, NODES * 4);
        fill_f32_kernel<<<(NODES * 4 + 255) / 256, 256, 0, stream>>>(DEN, 0.0f, NODES * 4);
        fill_f32_kernel<<<(aggN + 255) / 256, 256, 0, stream>>>(AGG, 0.0f, aggN);

        edge_alpha_kernel<<<ebk, 256, 0, stream>>>(Q, Kb, EA, We, src, dst, ALPH, AMAX, EDGES);
        edge_exp_kernel  <<<ebk, 256, 0, stream>>>(ALPH, AMAX, dst, DEN, EDGES);
        edge_msg_kernel  <<<ebk, 256, 0, stream>>>(Vb, EA, We, ALPH, DEN, src, dst, AGG, EDGES);

        node_merge_ln_kernel<<<nwbk, 256, 0, stream>>>(X, AGG, SKIP, F(p + 11), F(p + 12), NODES);
        launch_gemm(X, F(p + 9), F(p + 10), T1, NODES, 64, 64, 0, stream);     // Wlin,blin
        node_ln_add_kernel<<<nwbk, 256, 0, stream>>>(X, T1, F(p + 13), F(p + 14), NODES);
    }

    // ---- charge head -> out[2..]
    launch_gemm(X,  F(38), F(39), T1, NODES, 64, 64, 1, stream);
    launch_gemm(T1, F(40), F(41), T2, NODES, 64, 64, 1, stream);
    gemv64_kernel<<<(NODES * 32 + 255) / 256, 256, 0, stream>>>(T2, F(42), F(43), out + 2, NODES);

    // ---- G / gap heads -> out[0], out[1]
    fill_f32_kernel<<<1, 32, 0, stream>>>(out, 0.0f, 2);

    launch_gemm(X,  F(44), F(45), T1, NODES, 64, 64, 1, stream);
    launch_gemm(T1, F(46), F(47), T2, NODES, 64, 64, 1, stream);
    gemv64_kernel<<<(NODES * 32 + 255) / 256, 256, 0, stream>>>(T2, F(48), F(49), GB, NODES);
    reduce_sum_kernel<<<64, 256, 0, stream>>>(GB, out + 0, NODES);

    launch_gemm(X,  F(50), F(51), T1, NODES, 64, 64, 1, stream);
    launch_gemm(T1, F(52), F(53), T2, NODES, 64, 64, 1, stream);
    gemv64_kernel<<<(NODES * 32 + 255) / 256, 256, 0, stream>>>(T2, F(54), F(55), GB, NODES);
    reduce_sum_kernel<<<64, 256, 0, stream>>>(GB, out + 1, NODES);

    (void)in_sizes; (void)n_in; (void)out_size; (void)ws_size;
}
